// ScaledDotProductAttention_35751307772254
// MI455X (gfx1250) — matrix operand order repaired and checked by reading, not compiled, not run on hardware
//
#include <hip/hip_runtime.h>
#include <math.h>

typedef __attribute__((ext_vector_type(16))) _Float16 v16h;
typedef __attribute__((ext_vector_type(8)))  _Float16 v8h;
typedef __attribute__((ext_vector_type(4)))  _Float16 v4h;
typedef __attribute__((ext_vector_type(8)))  float    v8f;
typedef __attribute__((ext_vector_type(4)))  float    v4f;

#define BATCH 4
#define HEADS 16
#define SEQ   2048
#define DKK   64
#define QTILE 128      // query rows per workgroup (8 waves x 16)
#define KVT   32       // kv block size
#define KPAD  72       // K LDS row stride in halves (64 + 8)  -> conflict-free b128 reads
#define VPAD  40       // V^T LDS row stride in halves (32 + 8) -> conflict-free b128 reads

__device__ __forceinline__ unsigned pack2(float a, float b) {
    union { _Float16 h[2]; unsigned u; } x;
    x.h[0] = (_Float16)a; x.h[1] = (_Float16)b;
    return x.u;
}

// Issue next-block global loads (f32 kept raw in regs; converted after compute).
__device__ __forceinline__ void load_regs(const float* __restrict__ Kp,
                                          const float* __restrict__ Vp,
                                          int kb, int tid,
                                          float kr[8], float vr[4][2]) {
    const int rk = tid >> 3;            // kv row 0..31
    const int ck = (tid & 7) * 8;       // col 0..56
    const float* ks = Kp + (size_t)(kb + rk) * DKK + ck;
    #pragma unroll
    for (int j = 0; j < 8; ++j) kr[j] = ks[j];

    const int r4  = (tid >> 5) * 4;     // kv rows r4..r4+3
    const int c0v = (tid & 31) * 2;     // cols c0v, c0v+1
    #pragma unroll
    for (int i = 0; i < 4; ++i) {
        const float* vs = Vp + (size_t)(kb + r4 + i) * DKK + c0v;
        vr[i][0] = vs[0];
        vr[i][1] = vs[1];
    }
}

// Convert f32->f16 and commit to LDS buffer: K row-major (b128), V^T packed (b64).
__device__ __forceinline__ void store_lds(_Float16* __restrict__ Kb,
                                          _Float16* __restrict__ Vb,
                                          int tid,
                                          const float kr[8], const float vr[4][2]) {
    const int rk = tid >> 3;
    const int ck = (tid & 7) * 8;
    v8h kh;
    #pragma unroll
    for (int j = 0; j < 8; ++j) kh[j] = (_Float16)kr[j];
    *(v8h*)&Kb[rk * KPAD + ck] = kh;

    const int r4  = (tid >> 5) * 4;
    const int c0v = (tid & 31) * 2;
    #pragma unroll
    for (int j = 0; j < 2; ++j) {
        v4h pv;
        #pragma unroll
        for (int i = 0; i < 4; ++i) pv[i] = (_Float16)vr[i][j];
        *(v4h*)&Vb[(c0v + j) * VPAD + r4] = pv;   // Vt[d][kv], 8B aligned
    }
}

__global__ __launch_bounds__(256, 2)
void fa_fwd_causal(const float* __restrict__ Q, const float* __restrict__ K,
                   const float* __restrict__ V, float* __restrict__ O) {
    __shared__ __align__(16) _Float16 Kbuf[2][KVT * KPAD];  // K block, row-major f16
    __shared__ __align__(16) _Float16 Vbuf[2][DKK * VPAD];  // V block transposed f16

    const int tid  = threadIdx.x;
    const int wave = tid >> 5;
    const int lane = tid & 31;
    const int hlf  = lane >> 4;    // 0 or 1 (lane half)
    const int l16  = lane & 15;

    const int bh    = blockIdx.y;
    const int q0blk = blockIdx.x * QTILE;
    const int q0    = q0blk + wave * 16;   // this wave's 16 query rows

    const float* Qp = Q + (size_t)bh * SEQ * DKK;
    const float* Kp = K + (size_t)bh * SEQ * DKK;
    const float* Vp = V + (size_t)bh * SEQ * DKK;
    float*       Op = O + (size_t)bh * SEQ * DKK;

    // ---- Q^T B-fragments (lane holds Q[m=l16][dbase + 16*hlf + j]), scale folded in
    const float qscale = 0.125f * 1.44269504088896340736f;  // 1/sqrt(64) * log2(e)
    v16h qf[2];
    {
        const float* qr = Qp + (size_t)(q0 + l16) * DKK;
        #pragma unroll
        for (int db = 0; db < 2; ++db) {
            const float* src = qr + db * 32 + hlf * 16;
            #pragma unroll
            for (int j = 0; j < 16; ++j) qf[db][j] = (_Float16)(src[j] * qscale);
        }
    }

    v8f acc[4];                      // O^T tiles: acc[dt][r] = O[m=l16][d=16*dt+8*hlf+r]
    #pragma unroll
    for (int i = 0; i < 4; ++i)
        #pragma unroll
        for (int r = 0; r < 8; ++r) acc[i][r] = 0.0f;
    float mrun = -INFINITY;
    float lrun = 0.0f;

    const int kb_end = q0blk + QTILE;          // causal extent of this query tile

    // ---- prologue: stage block 0
    float kr[8], vr[4][2];
    load_regs(Kp, Vp, 0, tid, kr, vr);
    store_lds(&Kbuf[0][0], &Vbuf[0][0], tid, kr, vr);
    __syncthreads();

    int pbuf = 0;
    for (int kb = 0; kb < kb_end; kb += KVT) {
        const bool has_next = (kb + KVT) < kb_end;
        if (has_next) load_regs(Kp, Vp, kb + KVT, tid, kr, vr);  // in flight over compute

        if (kb < q0 + 16) {                    // wave-uniform causal skip
            const _Float16* Kl = &Kbuf[pbuf][0];
            const _Float16* Vl = &Vbuf[pbuf][0];
            // ---- S^T = K x Q^T : two 16-kv tiles, k-dim 64 = 2 wmma each
            v8f st[2];
            #pragma unroll
            for (int t = 0; t < 2; ++t) {
                v8f c;
                #pragma unroll
                for (int r = 0; r < 8; ++r) c[r] = 0.0f;
                #pragma unroll
                for (int db = 0; db < 2; ++db) {
                    const _Float16* base = &Kl[(t * 16 + l16) * KPAD + db * 32 + hlf * 16];
                    v8h lo = *(const v8h*)base;
                    v8h hi = *(const v8h*)(base + 8);
                    v16h a;
                    #pragma unroll
                    for (int j = 0; j < 8; ++j) { a[j] = lo[j]; a[j + 8] = hi[j]; }
                    c = __builtin_amdgcn_wmma_f32_16x16x32_f16(false, a, false, qf[db],
                                                               (short)0, c, false, false);
                }
                st[t] = c;
            }
            // ---- causal mask on diagonal blocks (lane holds kv = kb+16t+8*hlf+r, m = q0+l16)
            const int mrow = q0 + l16;
            if (kb + KVT > q0) {
                #pragma unroll
                for (int t = 0; t < 2; ++t)
                    #pragma unroll
                    for (int r = 0; r < 8; ++r) {
                        const int kv = kb + t * 16 + hlf * 8 + r;
                        if (kv > mrow) st[t][r] = -INFINITY;
                    }
            }
            // ---- online softmax: row stats need only a lane<->lane^16 exchange
            float bmax = st[0][0];
            #pragma unroll
            for (int r = 1; r < 8; ++r) bmax = fmaxf(bmax, st[0][r]);
            #pragma unroll
            for (int r = 0; r < 8; ++r) bmax = fmaxf(bmax, st[1][r]);
            bmax = fmaxf(bmax, __shfl_xor(bmax, 16, 32));
            const float mnew  = fmaxf(mrun, bmax);
            const float alpha = __builtin_amdgcn_exp2f(mrun - mnew);
            mrun = mnew;
            float p[2][8];
            float rsum = 0.0f;
            #pragma unroll
            for (int t = 0; t < 2; ++t)
                #pragma unroll
                for (int r = 0; r < 8; ++r) {
                    const float e = __builtin_amdgcn_exp2f(st[t][r] - mnew);
                    p[t][r] = e; rsum += e;
                }
            rsum += __shfl_xor(rsum, 16, 32);
            lrun = lrun * alpha + rsum;
            #pragma unroll
            for (int i = 0; i < 4; ++i)
                #pragma unroll
                for (int r = 0; r < 8; ++r) acc[i][r] *= alpha;

            // ---- build P^T B-fragment (halves j <-> kv = 16*hlf + j) via xor-16 exchange
            unsigned own[2][4], oth[2][4];
            #pragma unroll
            for (int t = 0; t < 2; ++t)
                #pragma unroll
                for (int j = 0; j < 4; ++j) {
                    own[t][j] = pack2(p[t][2 * j], p[t][2 * j + 1]);
                    oth[t][j] = __shfl_xor(own[t][j], 16, 32);
                }
            union { unsigned u[8]; v16h v; } pf;
            #pragma unroll
            for (int j = 0; j < 4; ++j) {
                pf.u[j]     = hlf ? oth[1][j] : own[0][j];   // kv 16..23 | kv 0..7
                pf.u[j + 4] = hlf ? own[1][j] : oth[0][j];   // kv 24..31 | kv 8..15
            }
            // ---- O^T += V^T x P^T : 4 d-tiles, one wmma each (k-dim = 32 kv)
            #pragma unroll
            for (int dt = 0; dt < 4; ++dt) {
                const _Float16* base = &Vl[(dt * 16 + l16) * VPAD + hlf * 16];
                v8h lo = *(const v8h*)base;
                v8h hi = *(const v8h*)(base + 8);
                v16h a;
                #pragma unroll
                for (int j = 0; j < 8; ++j) { a[j] = lo[j]; a[j + 8] = hi[j]; }
                acc[dt] = __builtin_amdgcn_wmma_f32_16x16x32_f16(false, a, false, pf.v,
                                                                 (short)0, acc[dt], false, false);
            }
        }

        if (has_next)
            store_lds(&Kbuf[pbuf ^ 1][0], &Vbuf[pbuf ^ 1][0], tid, kr, vr);
        __syncthreads();
        pbuf ^= 1;
    }

    // ---- epilogue: O[m][d] = acc/l, 16B vector stores (d runs contiguously in r)
    const float rl = 1.0f / lrun;
    float* orow = Op + (size_t)(q0 + l16) * DKK + hlf * 8;
    #pragma unroll
    for (int dt = 0; dt < 4; ++dt) {
        v4f v0, v1;
        #pragma unroll
        for (int r = 0; r < 4; ++r) { v0[r] = acc[dt][r] * rl; v1[r] = acc[dt][r + 4] * rl; }
        *(v4f*)(orow + dt * 16)     = v0;
        *(v4f*)(orow + dt * 16 + 4) = v1;
    }
}

extern "C" void kernel_launch(void* const* d_in, const int* in_sizes, int n_in,
                              void* d_out, int out_size, void* d_ws, size_t ws_size,
                              hipStream_t stream) {
    const float* q = (const float*)d_in[0];
    const float* k = (const float*)d_in[1];
    const float* v = (const float*)d_in[2];
    // d_in[3] = causal mask (bool) — causality is computed analytically in-kernel.
    float* o = (float*)d_out;
    dim3 grid(SEQ / QTILE, BATCH * HEADS);
    dim3 block(256);
    hipLaunchKernelGGL(fa_fwd_causal, grid, block, 0, stream, q, k, v, o);
}
